// SimpleCnnNcpActor_39943195853201
// MI455X (gfx1250) — compile-verified, matrix operand order checked by reading, not run on hardware
//
#include <hip/hip_runtime.h>
#include <hip/hip_bf16.h>

typedef __attribute__((ext_vector_type(16))) _Float16 v16h;
typedef __attribute__((ext_vector_type(8)))  float    v8f;

union Frag16 { v16h v; unsigned u[8]; };

// ---------------------------------------------------------------------------
// Implicit-GEMM conv (3x3, stride 2, VALID) + bias + ReLU using WMMA f16->f32.
// Block = 256 threads = 8 waves laid out as MT x NT tiles of 16x16.
// A = im2col(input)  [M x Kred],  B = weights^T [Kred x COUT], D = A*B.
// ---------------------------------------------------------------------------
template<int CIN, int COUT, int HIN, int WIN, int HOUT, int WOUT, int MT, int NT>
__global__ __launch_bounds__(256)
void conv_wmma_kernel(const float* __restrict__ in, const float* __restrict__ wgt,
                      const float* __restrict__ bias, float* __restrict__ out, int nimg)
{
    constexpr int KRED = CIN * 9;
    constexpr int KCH  = (KRED + 31) / 32;
    constexpr int TM   = MT * 16;
    constexpr int TN   = NT * 16;   // == COUT for all three convs
    constexpr int AS   = 40;        // padded f16 row stride (bank-conflict break)

    __shared__ __align__(16) _Float16 Alds[TM][AS];
    __shared__ __align__(16) _Float16 Blds[TN][AS];   // stored transposed: [n][k]

    const int tid    = threadIdx.x;
    const int Mtot   = nimg * HOUT * WOUT;
    const int mblock = blockIdx.x * TM;

    const int lane = tid & 31;
    const int wv   = tid >> 5;       // wave 0..7
    const int wm   = wv / NT;        // row tile within block
    const int wn   = wv % NT;        // col tile within block
    const int r16  = lane & 15;
    const int half = lane >> 4;      // 0: K 0-7/16-23, 1: K 8-15/24-31

    v8f acc = {};

    for (int kc = 0; kc < KCH; ++kc) {
        __syncthreads();
        // ---- stage A tile (im2col gather, fp32 -> f16) ----
        for (int idx = tid; idx < TM * 32; idx += 256) {
            int ml = idx >> 5;
            int kl = idx & 31;
            int k  = kc * 32 + kl;
            int m  = mblock + ml;
            float val = 0.f;
            if (k < KRED && m < Mtot) {
                int c = k / 9, rs = k % 9, r = rs / 3, s = rs % 3;
                int img = m / (HOUT * WOUT);
                int pq  = m % (HOUT * WOUT);
                int p = pq / WOUT, q = pq % WOUT;
                val = in[((img * CIN + c) * HIN + (p * 2 + r)) * WIN + (q * 2 + s)];
            }
            Alds[ml][kl] = (_Float16)val;
        }
        // ---- stage B tile transposed: Blds[n][k] = w[n][c][r][s] ----
        for (int idx = tid; idx < TN * 32; idx += 256) {
            int nl = idx >> 5;
            int kl = idx & 31;
            int k  = kc * 32 + kl;
            float val = 0.f;
            if (k < KRED) {
                int c = k / 9, rs = k % 9, r = rs / 3, s = rs % 3;
                val = wgt[((nl * CIN + c) * 3 + r) * 3 + s];
            }
            Blds[nl][kl] = (_Float16)val;
        }
        __syncthreads();

        // ---- gather ISA-layout fragments (K pairs contiguous -> 32b reads) ----
        Frag16 A, Bf;
        const _Float16* ar = &Alds[wm * 16 + r16][0];
        const _Float16* br = &Blds[wn * 16 + r16][0];
        const int kb = half * 8;
        #pragma unroll
        for (int j = 0; j < 4; ++j) {
            A.u[j]      = *(const unsigned*)(ar + kb + 2 * j);
            A.u[4 + j]  = *(const unsigned*)(ar + 16 + kb + 2 * j);
            Bf.u[j]     = *(const unsigned*)(br + kb + 2 * j);
            Bf.u[4 + j] = *(const unsigned*)(br + 16 + kb + 2 * j);
        }
        acc = __builtin_amdgcn_wmma_f32_16x16x32_f16(false, A.v, false, Bf.v,
                                                     (short)0, acc, false, false);
    }

    // ---- epilogue: bias + ReLU, write NCHW ----
    const int   n     = wn * 16 + r16;
    const float bv    = bias[n];
    const int   mbase = mblock + wm * 16 + half * 8;
    #pragma unroll
    for (int r = 0; r < 8; ++r) {
        int m = mbase + r;
        if (m < Mtot) {
            int img = m / (HOUT * WOUT);
            int pq  = m % (HOUT * WOUT);
            int p = pq / WOUT, q = pq % WOUT;
            float v = acc[r] + bv;
            out[((img * COUT + n) * HOUT + p) * WOUT + q] = v > 0.f ? v : 0.f;
        }
    }
}

// ---------------------------------------------------------------------------
// Sensory synapse sums: independent of LTC state -> fully parallel over (b,t).
// 8 (b,t) pairs per block so each parameter read is amortized over 8 sigmoids
// (cuts the dominant L2 parameter traffic 8x). Fuses concat + input affine.
// ---------------------------------------------------------------------------
__global__ __launch_bounds__(256)
void sensory_kernel(const float* __restrict__ feat, const float* __restrict__ relpos,
                    const float* __restrict__ in_w, const float* __restrict__ in_b,
                    const float* __restrict__ s_sigma, const float* __restrict__ s_mu,
                    const float* __restrict__ s_w, const float* __restrict__ s_erev,
                    const int* __restrict__ sens_mask,
                    float* __restrict__ wnum, float* __restrict__ wden)
{
    const int bt0 = blockIdx.x * 8;
    const int tid = threadIdx.x;
    __shared__ float x[8][1283];            // ~41 KB of the 320 KB WGP LDS
    __shared__ float pn[5][48];
    __shared__ float pd[5][48];

    for (int idx = tid; idx < 8 * 1283; idx += 256) {
        int u = idx / 1283, i = idx % 1283;
        int bt = bt0 + u;
        float raw = (i < 1280) ? feat[(size_t)bt * 1280 + i] : relpos[bt * 3 + (i - 1280)];
        x[u][i] = raw * in_w[i] + in_b[i];
    }
    __syncthreads();

    const int j = tid % 48, g = tid / 48;
    float an[8] = {}, ad[8] = {};
    if (tid < 240) {                        // 5 groups x 48 posts
        for (int i = g; i < 1283; i += 5) {
            int idx = i * 48 + j;
            float sg = s_sigma[idx];
            float m_ = s_mu[idx];
            float w_ = s_w[idx] * (float)sens_mask[idx];   // fold mask
            float er = s_erev[idx];
            #pragma unroll
            for (int u = 0; u < 8; ++u) {
                float z  = (x[u][i] - m_) * sg;
                float sa = w_ * (1.f / (1.f + __expf(-z)));
                an[u] += sa * er;
                ad[u] += sa;
            }
        }
    }
    #pragma unroll
    for (int u = 0; u < 8; ++u) {
        if (tid < 240) { pn[g][j] = an[u]; pd[g][j] = ad[u]; }
        __syncthreads();
        if (tid < 48) {
            float n_ = 0.f, d_ = 0.f;
            #pragma unroll
            for (int gg = 0; gg < 5; ++gg) { n_ += pn[gg][tid]; d_ += pd[gg][tid]; }
            wnum[(bt0 + u) * 48 + tid] = n_;
            wden[(bt0 + u) * 48 + tid] = d_;
        }
        __syncthreads();
    }
}

// ---------------------------------------------------------------------------
// Sequential LTC scan (1 block per batch element) + pooled head + tanh.
// 48x48 synapse params staged in LDS once; all 288 unfolds hit LDS instead of
// re-streaming ~46 KB from L2 per unfold (latency-bound inner loop).
// ---------------------------------------------------------------------------
__global__ __launch_bounds__(256)
void ltc_scan_kernel(const float* __restrict__ wnum, const float* __restrict__ wden,
                     const float* __restrict__ gleak, const float* __restrict__ vleak,
                     const float* __restrict__ cm, const float* __restrict__ sigma,
                     const float* __restrict__ mu, const float* __restrict__ wsyn,
                     const float* __restrict__ erev, const int* __restrict__ syn_mask,
                     const float* __restrict__ w_out, const float* __restrict__ b_out,
                     const float* __restrict__ w_head, const float* __restrict__ b_head,
                     float* __restrict__ out)
{
    const int b   = blockIdx.x;
    const int tid = threadIdx.x;
    __shared__ float sigL[2304], muL[2304], wmL[2304], erL[2304];   // ~37 KB
    __shared__ float v[48];
    __shared__ float pn[5][48];
    __shared__ float pd[5][48];
    __shared__ float vsum[4];

    for (int i = tid; i < 2304; i += 256) {
        sigL[i] = sigma[i];
        muL[i]  = mu[i];
        wmL[i]  = wsyn[i] * (float)syn_mask[i];    // fold sparsity mask
        erL[i]  = erev[i];
    }
    if (tid < 48) v[tid] = 0.f;
    if (tid < 4)  vsum[tid] = 0.f;

    float gl = 0.f, glvl = 0.f, cmt = 0.f;
    if (tid < 48) {
        gl   = gleak[tid];
        glvl = gl * vleak[tid];
        cmt  = cm[tid] * 6.f;                      // cm * ODE_UNFOLDS
    }
    __syncthreads();

    const int j = tid % 48, g = tid / 48;
    for (int t = 0; t < 48; ++t) {
        const int base = (b * 48 + t) * 48;
        for (int u = 0; u < 6; ++u) {
            if (tid < 240) {
                float an = 0.f, ad = 0.f;
                for (int i = g; i < 48; i += 5) {
                    int idx = i * 48 + j;
                    float z  = (v[i] - muL[idx]) * sigL[idx];
                    float wa = wmL[idx] * (1.f / (1.f + __expf(-z)));
                    an += wa * erL[idx];
                    ad += wa;
                }
                pn[g][j] = an; pd[g][j] = ad;
            }
            __syncthreads();
            if (tid < 48) {
                float an = 0.f, ad = 0.f;
                #pragma unroll
                for (int gg = 0; gg < 5; ++gg) { an += pn[gg][tid]; ad += pd[gg][tid]; }
                float num = cmt * v[tid] + glvl + an + wnum[base + tid];
                float den = cmt + gl + ad + wden[base + tid] + 1e-8f;
                v[tid] = num / den;
            }
            __syncthreads();
        }
        if (tid < 4) vsum[tid] += v[tid];
        __syncthreads();
    }

    if (tid < 2) {  // pooled motor affine + Linear(4->2) + tanh
        float a = b_head[tid];
        #pragma unroll
        for (int jj = 0; jj < 4; ++jj) {
            float pooled = vsum[jj] * w_out[jj] * (1.f / 48.f) + b_out[jj];
            a += pooled * w_head[jj * 2 + tid];
        }
        out[b * 2 + tid] = tanhf(a);
    }
}

// ---------------------------------------------------------------------------
extern "C" void kernel_launch(void* const* d_in, const int* in_sizes, int n_in,
                              void* d_out, int out_size, void* d_ws, size_t ws_size,
                              hipStream_t stream)
{
    const float* image   = (const float*)d_in[0];
    const float* relpos  = (const float*)d_in[1];
    const float* w0      = (const float*)d_in[2];
    const float* b0      = (const float*)d_in[3];
    const float* w1      = (const float*)d_in[4];
    const float* b1      = (const float*)d_in[5];
    const float* w2      = (const float*)d_in[6];
    const float* b2      = (const float*)d_in[7];
    const float* in_w    = (const float*)d_in[8];
    const float* in_b    = (const float*)d_in[9];
    const float* gleak   = (const float*)d_in[10];
    const float* vleak   = (const float*)d_in[11];
    const float* cm      = (const float*)d_in[12];
    const float* sigma   = (const float*)d_in[13];
    const float* mu      = (const float*)d_in[14];
    const float* wsyn    = (const float*)d_in[15];
    const float* erev    = (const float*)d_in[16];
    const float* s_sigma = (const float*)d_in[17];
    const float* s_mu    = (const float*)d_in[18];
    const float* s_w     = (const float*)d_in[19];
    const float* s_erev  = (const float*)d_in[20];
    const float* w_out   = (const float*)d_in[21];
    const float* b_out   = (const float*)d_in[22];
    const float* w_head  = (const float*)d_in[23];
    const float* b_head  = (const float*)d_in[24];
    const int*   syn_mask  = (const int*)d_in[25];
    const int*   sens_mask = (const int*)d_in[26];

    const int NIMG = 32 * 48;   // B*S = 1536

    float* ws   = (float*)d_ws;
    float* out1 = ws;                                     // [1536,32,23,35]
    float* out2 = out1 + (size_t)NIMG * 32 * 23 * 35;     // [1536,64,11,17]
    float* out3 = out2 + (size_t)NIMG * 64 * 11 * 17;     // [1536,1280]
    float* wnum = out3 + (size_t)NIMG * 1280;             // [1536,48]
    float* wden = wnum + (size_t)NIMG * 48;               // [1536,48]

    // conv1: 1x48x72 -> 32x23x35   (Kred=9 padded to 32; MTxNT = 4x2)
    {
        int M = NIMG * 23 * 35;
        conv_wmma_kernel<1, 32, 48, 72, 23, 35, 4, 2>
            <<<(M + 63) / 64, 256, 0, stream>>>(image, w0, b0, out1, NIMG);
    }
    // conv2: 32x23x35 -> 64x11x17  (Kred=288 = 9 chunks; MTxNT = 2x4)
    {
        int M = NIMG * 11 * 17;
        conv_wmma_kernel<32, 64, 23, 35, 11, 17, 2, 4>
            <<<(M + 31) / 32, 256, 0, stream>>>(out1, w1, b1, out2, NIMG);
    }
    // conv3: 64x11x17 -> 32x5x8    (Kred=576 = 18 chunks; MTxNT = 4x2)
    {
        int M = NIMG * 5 * 8;
        conv_wmma_kernel<64, 32, 11, 17, 5, 8, 4, 2>
            <<<(M + 63) / 64, 256, 0, stream>>>(out2, w2, b2, out3, NIMG);
    }

    // state-independent sensory synapse sums, 8 (b,t) pairs per block
    sensory_kernel<<<NIMG / 8, 256, 0, stream>>>(out3, relpos, in_w, in_b,
                                                 s_sigma, s_mu, s_w, s_erev, sens_mask,
                                                 wnum, wden);

    // sequential ODE scan + head
    ltc_scan_kernel<<<32, 256, 0, stream>>>(wnum, wden, gleak, vleak, cm,
                                            sigma, mu, wsyn, erev, syn_mask,
                                            w_out, b_out, w_head, b_head,
                                            (float*)d_out);
}
